// SemSol_8830452761095
// MI455X (gfx1250) — matrix operand order repaired and checked by reading, not compiled, not run on hardware
//
#include <hip/hip_runtime.h>
#include <hip/hip_bf16.h>
#include <stdint.h>

// ---- problem constants -----------------------------------------------------
#define Bk   8
#define Sk   2048
#define Hk   768
#define HDk  384
#define G4   1536            // 4*HD gate width
#define NTn  96              // N tiles (1536/16)
#define KTih 24              // K tiles for input proj (768/32)
#define KThh 12              // K tiles for recurrence (384/32)
#define NIH  (KTih*NTn*256)  // packed Wih dwords per dir  = 589824
#define NHH  (KThh*NTn*256)  // packed Whh dwords per dir  = 294912
#define HXW  (KThh*256)      // h fragment dwords per (dir,phase) = 3072

typedef __attribute__((ext_vector_type(16))) __bf16 v16bf;
typedef __attribute__((ext_vector_type(8)))  float  v8f;
typedef __attribute__((ext_vector_type(8)))  uint32_t v8u;

// ---- helpers ---------------------------------------------------------------
__device__ __forceinline__ uint16_t f2bf(float f) {
  uint32_t u = __builtin_bit_cast(uint32_t, f);
  u += 0x7FFFu + ((u >> 16) & 1u);          // round-to-nearest-even
  return (uint16_t)(u >> 16);
}
__device__ __forceinline__ uint32_t pack2(float a, float b) {
  return (uint32_t)f2bf(a) | ((uint32_t)f2bf(b) << 16);
}
// A-fragment (16x32 bf16): lane m=l&15; per-VGPR K start
__device__ __forceinline__ int a_k0(int lane, int v) {
  int base = (v < 4) ? 2 * v : 16 + 2 * (v - 4);
  return base + ((lane & 16) ? 8 : 0);
}
__device__ __forceinline__ float sigf(float x) { return 1.0f / (1.0f + expf(-x)); }

__device__ __forceinline__ v8u ld8_g(const uint32_t* p) {
  uint4 a = *(const uint4*)p, b = *(const uint4*)(p + 4);
  v8u r = {a.x, a.y, a.z, a.w, b.x, b.y, b.z, b.w};
  return r;
}

__device__ __forceinline__ v8f wmma_bf16(v8u a, v8u b, v8f c) {
  return __builtin_amdgcn_wmma_f32_16x16x32_bf16(
      false, __builtin_bit_cast(v16bf, a),
      false, __builtin_bit_cast(v16bf, b),
      (short)0, c, false, false);
}

// async global->LDS copy (ASYNCcnt path), 16B per lane
__device__ __forceinline__ void async_cp16(uint32_t lds_off, const void* gptr) {
  asm volatile("global_load_async_to_lds_b128 %0, %1, off"
               :: "v"(lds_off), "v"((uint64_t)(uintptr_t)gptr) : "memory");
}

// ---- kernel 0: zero the cross-block sync counters (graph-replay safe) ------
__global__ void k_init(uint32_t* cnt) {
  if (threadIdx.x < 2) cnt[threadIdx.x] = 0u;
}

// ---- kernel 1: pack weights into WMMA B-fragment layout, combine biases ----
__global__ void k_pack(const float* WihF, const float* WhhF,
                       const float* bihF, const float* bhhF,
                       const float* WihB, const float* WhhB,
                       const float* bihB, const float* bhhB,
                       uint32_t* WihP, uint32_t* WhhP, float* biasC) {
  size_t stride = (size_t)gridDim.x * blockDim.x;
  size_t i0 = (size_t)blockIdx.x * blockDim.x + threadIdx.x;
  for (size_t idx = i0; idx < (size_t)2 * NIH; idx += stride) {
    int dir = (int)(idx / NIH); size_t r = idx % NIH;
    int kt = (int)(r / (NTn * 256)); int rem = (int)(r % (NTn * 256));
    int nt = rem >> 8; int l = (rem & 255) >> 3; int v = rem & 7;
    const float* W = dir ? WihB : WihF;
    int n = nt * 16 + (l & 15);
    int k = kt * 32 + a_k0(l, v);
    WihP[idx] = pack2(W[n * Hk + k], W[n * Hk + k + 1]);
  }
  for (size_t idx = i0; idx < (size_t)2 * NHH; idx += stride) {
    int dir = (int)(idx / NHH); size_t r = idx % NHH;
    int kt = (int)(r / (NTn * 256)); int rem = (int)(r % (NTn * 256));
    int nt = rem >> 8; int l = (rem & 255) >> 3; int v = rem & 7;
    const float* W = dir ? WhhB : WhhF;
    int n = nt * 16 + (l & 15);
    int k = kt * 32 + a_k0(l, v);
    WhhP[idx] = pack2(W[n * HDk + k], W[n * HDk + k + 1]);
  }
  for (size_t idx = i0; idx < (size_t)2 * G4; idx += stride) {
    int dir = (int)(idx / G4); int n = (int)(idx % G4);
    biasC[idx] = dir ? (bihB[n] + bhhB[n]) : (bihF[n] + bhhF[n]);
  }
}

// ---- kernel 2: input projection, timestep-pair packed (M=16 fully used) ----
// A tile rows 0..7 = batch @ t=2*tau, rows 8..15 = batch @ t=2*tau+1.
// Output C-fragments: lanes 0-15 hold even t, lanes 16-31 hold odd t.
__global__ __launch_bounds__(256) void k_inproj(const float* __restrict__ X,
                                                const uint32_t* __restrict__ WihP,
                                                const float* __restrict__ biasC,
                                                float* __restrict__ Xp) {
  int tau = blockIdx.x, dir = blockIdx.y;
  __shared__ __align__(16) uint32_t aF[KTih * 256];
  int tid = threadIdx.x;
  for (int d = tid; d < KTih * 256; d += 256) {
    int kt = d >> 8; int r = d & 255; int l = r >> 3; int v = r & 7;
    int m = l & 15;
    int b = m & 7, tt = 2 * tau + (m >> 3);
    int k = kt * 32 + a_k0(l, v);
    const float* xr = X + ((size_t)b * Sk + tt) * Hk;
    aF[d] = pack2(xr[k], xr[k + 1]);
  }
  __syncthreads();
  int lane = tid & 31, w = tid >> 5;
  const uint32_t* Wp = WihP + (size_t)dir * NIH;
  for (int nn = 0; nn < 12; ++nn) {
    int nt = w + nn * 8;
    v8f acc = {};
#pragma unroll
    for (int kt = 0; kt < KTih; ++kt) {
      v8u au = ld8_g(&aF[kt * 256 + lane * 8]);
      v8u bu = ld8_g(Wp + (((size_t)kt * NTn + nt) * 32 + lane) * 8);
      acc = wmma_bf16(au, bu, acc);
    }
    int n = nt * 16 + (lane & 15);
    float bn = biasC[dir * G4 + n];
    float* op = Xp + (((size_t)dir * (Sk / 2) + tau) * NTn + nt) * 256 + lane * 8;
#pragma unroll
    for (int q = 0; q < 8; q++) op[q] = acc[q] + bn;
  }
}

// ---- kernel 3: cooperative LSTM scan ---------------------------------------
// 8 blocks = 2 dirs x 4 j-slice blocks; Whh B-fragments register-resident for
// the whole scan; per-step cross-block h exchange (12KB) staged once per block
// into LDS with device-scope loads, then consumed via ds_load_b128.
__device__ __forceinline__ void cell2(int e, float& c, const float* gl,
                                      uint16_t* hxw16, float* hidden,
                                      int dir, int g, int t) {
  int b = e / 96, jj = e % 96;
  float gi = sigf(gl[b * 384 + jj]);
  float gf = sigf(gl[b * 384 + 96 + jj]);
  float gg = tanhf(gl[b * 384 + 192 + jj]);
  float go = sigf(gl[b * 384 + 288 + jj]);
  c = gf * c + gi * gg;
  float h = go * tanhf(c);
  int j = g * 96 + jj;
  hidden[((size_t)b * Sk + t) * Hk + dir * HDk + j] = h;
  int kt = j >> 5, kk = j & 31;
  int l2 = b + ((kk & 8) ? 16 : 0);
  int v2 = ((kk & 16) ? 4 : 0) + ((kk & 7) >> 1);
  hxw16[(uint32_t)(kt * 256 + l2 * 8 + v2) * 2u + (kk & 1)] = f2bf(h);
}

__global__ __launch_bounds__(256, 1) void k_scan(const float* __restrict__ Xp,
                                                 const uint32_t* __restrict__ WhhP,
                                                 float* __restrict__ hidden,
                                                 uint32_t* __restrict__ hx,
                                                 uint32_t* __restrict__ cnt) {
  int dir = blockIdx.x >> 2, g = blockIdx.x & 3;
  int tid = threadIdx.x, lane = tid & 31, w = tid >> 5;
  __shared__ __align__(16) float gl[Bk * 384];   // this block's gate slice
  __shared__ __align__(16) uint32_t hA[HXW];     // staged h A-fragments (12KB)

  // this block's 24 N-tiles: for gate q, nt = 24q + 6g + o  (o in 0..5)
  int nts[3];
#pragma unroll
  for (int nn = 0; nn < 3; ++nn) {
    int li = w * 3 + nn;                 // 0..23
    nts[nn] = 24 * (li / 6) + 6 * g + (li % 6);
  }

  // register-resident Whh B-fragments for the entire scan (3 x 12 x 8 VGPRs)
  const uint32_t* Wp = WhhP + (size_t)dir * NHH;
  v8u Breg[3][12];
#pragma unroll
  for (int nn = 0; nn < 3; ++nn)
#pragma unroll
    for (int kt = 0; kt < KThh; ++kt)
      Breg[nn][kt] = ld8_g(Wp + (((size_t)kt * NTn + nts[nn]) * 32 + lane) * 8);

  // zero our slice of the phase-0 h buffer (K-tiles 3g..3g+2)
  uint32_t* hx0 = hx + (dir * 2 + 0) * HXW;
  for (int i = tid; i < 768; i += 256) hx0[3 * g * 256 + i] = 0u;

  float c0 = 0.f, c1 = 0.f, c2 = 0.f;
  uint32_t* mycnt = cnt + dir;

  __threadfence();
  __syncthreads();
  if (tid == 0) {
    __hip_atomic_fetch_add(mycnt, 1u, __ATOMIC_ACQ_REL, __HIP_MEMORY_SCOPE_AGENT);
    while (__hip_atomic_load(mycnt, __ATOMIC_ACQUIRE, __HIP_MEMORY_SCOPE_AGENT) < 4u)
      __builtin_amdgcn_s_sleep(1);
  }
  __syncthreads();

  for (int step = 0; step < Sk; ++step) {
    int t = dir ? (Sk - 1 - step) : step;
    int p = step & 1;
    const uint32_t* hxr = hx + (dir * 2 + p) * HXW;
    uint16_t* hxw16 = (uint16_t*)(hx + (dir * 2 + (1 - p)) * HXW);
    int tau = t >> 1, par = (t & 1) ? 16 : 0;
    const float* xp = Xp + ((size_t)dir * (Sk / 2) + tau) * (NTn * 256);
    if (step + 1 < Sk) {
      int tn = dir ? (Sk - 2 - step) : step + 1;
      __builtin_prefetch(
          Xp + (((size_t)dir * (Sk / 2) + (tn >> 1)) * NTn + nts[0]) * 256, 0, 0);
    }

    // stage full h (all 12 K-tiles) once per block, device-scope (L2-fresh)
    for (int i = tid; i < HXW; i += 256)
      hA[i] = __hip_atomic_load(hxr + i, __ATOMIC_RELAXED, __HIP_MEMORY_SCOPE_AGENT);
    __syncthreads();

#pragma unroll
    for (int nn = 0; nn < 3; ++nn) {
      int nt = nts[nn];
      v8f acc = {};
      if (lane < 16) {  // pick the lane half matching t's parity
        const float* ip = xp + nt * 256 + (lane + par) * 8;
        float4 x0 = *(const float4*)ip, x1 = *(const float4*)(ip + 4);
        acc[0] = x0.x; acc[1] = x0.y; acc[2] = x0.z; acc[3] = x0.w;
        acc[4] = x1.x; acc[5] = x1.y; acc[6] = x1.z; acc[7] = x1.w;
      }
#pragma unroll
      for (int kt = 0; kt < KThh; ++kt)
        acc = wmma_bf16(ld8_g(&hA[kt * 256 + lane * 8]), Breg[nn][kt], acc);
      if (lane < 16) {
        int n = nt * 16 + lane;
        int q4 = n / 384, jj = (n % 384) - 96 * g;  // jj in [0,96)
#pragma unroll
        for (int vv = 0; vv < 8; vv++) gl[vv * 384 + q4 * 96 + jj] = acc[vv];
      }
    }
    __syncthreads();

    cell2(tid * 3 + 0, c0, gl, hxw16, hidden, dir, g, t);
    cell2(tid * 3 + 1, c1, gl, hxw16, hidden, dir, g, t);
    cell2(tid * 3 + 2, c2, gl, hxw16, hidden, dir, g, t);

    __threadfence();
    __syncthreads();
    if (tid == 0) {
      uint32_t tgt = 4u * (uint32_t)(step + 2);
      __hip_atomic_fetch_add(mycnt, 1u, __ATOMIC_ACQ_REL, __HIP_MEMORY_SCOPE_AGENT);
      while (__hip_atomic_load(mycnt, __ATOMIC_ACQUIRE, __HIP_MEMORY_SCOPE_AGENT) < tgt)
        __builtin_amdgcn_s_sleep(1);
    }
    __syncthreads();
  }
}

// ---- kernel 4: per-utterance max pool, async-to-LDS double-buffered --------
__global__ __launch_bounds__(256) void k_pool(const float* __restrict__ hidden,
                                              const int* __restrict__ utter,
                                              float* __restrict__ mp) {
  int b = blockIdx.x;
  int cb = blockIdx.y * 256;
  int tid = threadIdx.x;
  __shared__ __align__(16) float tile[2][8 * 256];  // 2 x 8 timesteps x 256 cols
  __shared__ float ml[33 * 256];
  for (int u = 0; u < 33; ++u) ml[u * 256 + tid] = 0.f;  // max(.,0) clamp built in

  const int NTILE = Sk / 8;  // 256 chunks of 8 timesteps

  auto issue_tile = [&](int kb, int buf) {
    const float* src = hidden + ((size_t)b * Sk + kb * 8) * Hk + cb;
    uint32_t lbase = (uint32_t)(uintptr_t)(&tile[buf][0]);
#pragma unroll
    for (int q = 0; q < 2; ++q) {
      int idx = q * 256 + tid;           // 0..511  -> (row, float4-col)
      int r = idx >> 6, c4 = idx & 63;
      async_cp16(lbase + (uint32_t)(r * 256 + c4 * 4) * 4u,
                 (const void*)(src + (size_t)r * Hk + c4 * 4));
    }
  };

  issue_tile(0, 0);
  for (int kb = 0; kb < NTILE; ++kb) {
    int buf = kb & 1;
    if (kb + 1 < NTILE) {
      issue_tile(kb + 1, 1 - buf);
      asm volatile("s_wait_asynccnt 0x2" ::: "memory");  // oldest tile landed
    } else {
      asm volatile("s_wait_asynccnt 0x0" ::: "memory");
    }
    __syncthreads();
#pragma unroll
    for (int r = 0; r < 8; ++r) {
      int u = utter[b * Sk + kb * 8 + r];
      int seg = u > 0 ? u : 0;
      float v = tile[buf][r * 256 + tid];
      float* pm = &ml[seg * 256 + tid];
      *pm = fmaxf(*pm, v);
    }
    __syncthreads();
  }
  for (int u = 1; u <= 32; ++u)
    mp[((size_t)b * 32 + (u - 1)) * Hk + cb + tid] = ml[u * 256 + tid];
}

// ---- kernel 5: topic softmax routing (tiny GEMV pair) ----------------------
__global__ __launch_bounds__(256) void k_topic(const float* __restrict__ mp,
                                               const float* __restrict__ tw,
                                               const float* __restrict__ tb,
                                               const float* __restrict__ ttab,
                                               float* __restrict__ emb) {
  int b = blockIdx.x, u = blockIdx.y;
  const float* row = mp + ((size_t)b * 32 + u) * Hk;
  __shared__ float rs[Hk];
  __shared__ float pr[64];
  __shared__ float inv_s;
  int tid = threadIdx.x;
  for (int h = tid; h < Hk; h += 256) rs[h] = row[h];
  __syncthreads();
  if (tid < 64) {
    float s = tb[tid];
    const float* wr = tw + tid * Hk;
    for (int h = 0; h < Hk; ++h) s += rs[h] * wr[h];
    pr[tid] = s;
  }
  __syncthreads();
  if (tid == 0) {
    float mx = pr[0];
    for (int q = 1; q < 64; ++q) mx = fmaxf(mx, pr[q]);
    float sm = 0.f;
    for (int q = 0; q < 64; ++q) { float e = expf(pr[q] - mx); pr[q] = e; sm += e; }
    inv_s = 1.0f / sm;
  }
  __syncthreads();
  float inv = inv_s;
  for (int h = tid; h < Hk; h += 256) {
    float s = 0.f;
    for (int q = 0; q < 64; ++q) s += pr[q] * ttab[q * Hk + h];
    emb[((size_t)b * 32 + u) * Hk + h] = s * inv;
  }
}

// ---- kernel 6: scatter back ------------------------------------------------
__global__ void k_scatter(const float* __restrict__ emb,
                          const int* __restrict__ utter,
                          float* __restrict__ out, size_t nvec) {
  size_t i = (size_t)blockIdx.x * blockDim.x + threadIdx.x;
  if (i >= nvec) return;
  int cv = (int)(i % (Hk / 4));
  size_t bs = i / (Hk / 4);
  int u = utter[bs];
  float4 r = {0.f, 0.f, 0.f, 0.f};
  if (u != 0) {
    int idx = (u > 0 ? u : -u) - 1; if (idx > 31) idx = 31;
    size_t b = bs / Sk;
    float4 gv = ((const float4*)(emb + ((b * 32 + (size_t)idx) * Hk)))[cv];
    float sc = (u > 0) ? 1.f : 2.f;
    r.x = gv.x * sc; r.y = gv.y * sc; r.z = gv.z * sc; r.w = gv.w * sc;
  }
  ((float4*)out)[i] = r;
}

// ---- launcher --------------------------------------------------------------
extern "C" void kernel_launch(void* const* d_in, const int* in_sizes, int n_in,
                              void* d_out, int out_size, void* d_ws, size_t ws_size,
                              hipStream_t stream) {
  const float* X     = (const float*)d_in[0];
  const float* WihF  = (const float*)d_in[1];
  const float* WhhF  = (const float*)d_in[2];
  const float* bihF  = (const float*)d_in[3];
  const float* bhhF  = (const float*)d_in[4];
  const float* WihB  = (const float*)d_in[5];
  const float* WhhB  = (const float*)d_in[6];
  const float* bihB  = (const float*)d_in[7];
  const float* bhhB  = (const float*)d_in[8];
  const float* tw    = (const float*)d_in[9];
  const float* tb    = (const float*)d_in[10];
  const float* ttab  = (const float*)d_in[11];
  const int*   utter = (const int*)d_in[12];

  float* ws = (float*)d_ws;
  size_t o = 0;
  uint32_t* WihP  = (uint32_t*)(ws + o); o += (size_t)2 * NIH;
  uint32_t* WhhP  = (uint32_t*)(ws + o); o += (size_t)2 * NHH;
  float*    biasC = ws + o;              o += (size_t)2 * G4;
  float*    Xp    = ws + o;              o += (size_t)2 * (Sk / 2) * (NTn * 256);
  float*    hid   = ws + o;              o += (size_t)Bk * Sk * Hk;
  float*    mp    = ws + o;              o += (size_t)Bk * 32 * Hk;
  float*    emb   = ws + o;              o += (size_t)Bk * 32 * Hk;
  uint32_t* hx    = (uint32_t*)(ws + o); o += (size_t)4 * HXW;   // 2 dirs x 2 phases
  uint32_t* cnt   = (uint32_t*)(ws + o); o += 64;

  k_init<<<1, 64, 0, stream>>>(cnt);
  k_pack<<<512, 256, 0, stream>>>(WihF, WhhF, bihF, bhhF, WihB, WhhB, bihB, bhhB,
                                  WihP, WhhP, biasC);
  dim3 gp(Sk / 2, 2);
  k_inproj<<<gp, 256, 0, stream>>>(X, WihP, biasC, Xp);
  k_scan<<<8, 256, 0, stream>>>(Xp, WhhP, hid, hx, cnt);
  dim3 gq(Bk, Hk / 256);
  k_pool<<<gq, 256, 0, stream>>>(hid, utter, mp);
  dim3 gt(Bk, 32);
  k_topic<<<gt, 256, 0, stream>>>(mp, tw, tb, ttab, emb);
  size_t nvec = (size_t)Bk * Sk * (Hk / 4);
  k_scatter<<<(unsigned)((nvec + 255) / 256), 256, 0, stream>>>(emb, utter,
                                                                (float*)d_out, nvec);
}
